// AttentionDecoder_33612414058955
// MI455X (gfx1250) — compile-verified, matrix operand order checked by reading, not run on hardware
//
#include <hip/hip_runtime.h>

// ---------------- types ----------------
typedef __attribute__((ext_vector_type(16))) __bf16 v16bf;
typedef __attribute__((ext_vector_type(8)))  __bf16 v8bf;
typedef __attribute__((ext_vector_type(4)))  __bf16 v4bf;
typedef __attribute__((ext_vector_type(8)))  float  v8f;
typedef __attribute__((ext_vector_type(4)))  unsigned int u32x4;
typedef __attribute__((ext_vector_type(8)))  int i32x8;
typedef __attribute__((ext_vector_type(4)))  int i32x4;

#define NB   32      // batch
#define TENC 500     // encoder frames
#define DD   1024    // encoder dim == d_model
#define TDEC 100     // decoder steps
#define EE   256     // embedding dim
#define CC   8000    // vocab
#define DM   1024    // d_model

#if defined(__has_builtin)
#if __has_builtin(__builtin_amdgcn_tensor_load_to_lds)
#define HAVE_TDM 1
#endif
#endif
#if __has_include(<hip/amd_detail/amd_gfx1250_TDM.h>)
#define TDM_6ARG 1   // therock-10.0 headers ship only with the 6-arg toolchain
#endif

static __device__ __forceinline__ __bf16 f2bf(float f) {
    union { float f; unsigned u; } x; x.f = f;
    unsigned r = (x.u + 0x7FFFu + ((x.u >> 16) & 1u)) >> 16;
    union { unsigned short s; __bf16 b; } y; y.s = (unsigned short)r;
    return y.b;
}
static __device__ __forceinline__ float bf2f(__bf16 b) {
    union { unsigned short s; __bf16 b; } y; y.b = b;
    union { unsigned u; float f; } x; x.u = ((unsigned)y.s) << 16;
    return x.f;
}
static __device__ __forceinline__ v8f wmma_bf16(v16bf a, v16bf b, v8f c) {
    return __builtin_amdgcn_wmma_f32_16x16x32_bf16(false, a, false, b,
                                                   (short)0, c, false, false);
}

// ---------------- elementwise converts / gathers ----------------
__global__ __launch_bounds__(256) void cvt_f32_bf16(const float* __restrict__ in,
                                                    __bf16* __restrict__ out, int n) {
    int i = blockIdx.x * 256 + threadIdx.x;
    if (i < n) out[i] = f2bf(in[i]);
}

// yembb[(n*TDEC+t), e] = bf16(emb[y[n,t], e])
__global__ __launch_bounds__(256) void gather_emb(const float* __restrict__ emb,
                                                  const int* __restrict__ y,
                                                  __bf16* __restrict__ yembb) {
    int idx = blockIdx.x * 256 + threadIdx.x;      // [0, 3200*256)
    int r = idx >> 8, e = idx & 255;
    if (r < NB * TDEC) yembb[idx] = f2bf(emb[(size_t)y[r] * EE + e]);
}

// Pre-swizzle row-major f32 B[K,N] into per-lane WMMA bf16 fragments, grouped
// for 64-wide N blocks:  [nb64][kt][sub(0..3)][lane][16 elems]
// Fragment element e of lane L -> K = (v&3)*2+(e&1)+(L>>4)*8+(v>>2)*16 (v=e>>1),
//                                 N = (nb64*4+sub)*16 + (L&15)
__global__ __launch_bounds__(256) void pack_b(const float* __restrict__ W,
                                              __bf16* __restrict__ Bsw, int K, int N) {
    int idx = blockIdx.x * 256 + threadIdx.x;
    int ktiles = K >> 5, ntiles = N >> 4;
    int total = ktiles * ntiles * 32;
    if (idx >= total) return;
    int lane = idx & 31;
    int tmp = idx >> 5;                 // [0, ktiles*ntiles)
    int sub = tmp & 3;
    int kt  = (tmp >> 2) % ktiles;
    int nb  = (tmp >> 2) / ktiles;
    int nt  = nb * 4 + sub;
    int half = lane >> 4, n = lane & 15;
    __bf16* out = Bsw + (size_t)idx * 16;
#pragma unroll
    for (int e = 0; e < 16; ++e) {
        int v = e >> 1;
        int k = ((v & 3) * 2) + (e & 1) + half * 8 + ((v >> 2) * 16);
        out[e] = f2bf(W[(size_t)(kt * 32 + k) * N + (nt * 16 + n)]);
    }
}

// ---------------- register-blocked bf16 WMMA GEMM (f32 accumulate) ----------------
// C[M,N] = A[M,K] @ B[K,N] (+bias). One wave computes a 32x64 tile:
// 2 A-fragments x 4 B-fragments -> 8 v_wmma per K-step (0.75KB loaded / wmma).
// Requires M%32==0, N%64==0, K%32==0 (true for all GEMMs here).
template <bool OUT_BF16>
__global__ __launch_bounds__(256) void wmma_gemm_kernel(
    const __bf16* __restrict__ A, const __bf16* __restrict__ Bsw,
    const float* __restrict__ bias, void* __restrict__ C,
    int mblk, int nblk, int ktiles, int lda, int ldc) {
    int waveId = blockIdx.x * 8 + (threadIdx.x >> 5);
    if (waveId >= mblk * nblk) return;          // wave-uniform exit keeps EXEC all-1s
    int mb = waveId % mblk, nb = waveId / mblk;
    int lane = threadIdx.x & 31;
    int half = lane >> 4, l15 = lane & 15;

    const __bf16* Arow0 = A + (size_t)(mb * 32 + l15) * lda;
    const __bf16* Arow1 = Arow0 + (size_t)16 * lda;
    const __bf16* Bp = Bsw + (size_t)nb * ktiles * 2048 + (size_t)lane * 16;

    v8f acc[2][4] = {};
    for (int kt = 0; kt < ktiles; ++kt) {
        v8bf t0 = *(const v8bf*)(Arow0 + kt * 32 + half * 8);
        v8bf t1 = *(const v8bf*)(Arow0 + kt * 32 + 16 + half * 8);
        v8bf t2 = *(const v8bf*)(Arow1 + kt * 32 + half * 8);
        v8bf t3 = *(const v8bf*)(Arow1 + kt * 32 + 16 + half * 8);
        v16bf a0, a1;
#pragma unroll
        for (int i = 0; i < 8; ++i) {
            a0[i] = t0[i]; a0[i + 8] = t1[i];
            a1[i] = t2[i]; a1[i + 8] = t3[i];
        }
#pragma unroll
        for (int s = 0; s < 4; ++s) {
            v16bf b = *(const v16bf*)(Bp + s * 512);
            acc[0][s] = wmma_bf16(a0, b, acc[0][s]);
            acc[1][s] = wmma_bf16(a1, b, acc[1][s]);
        }
        Bp += 2048;
    }
#pragma unroll
    for (int s = 0; s < 4; ++s) {
        int col = nb * 64 + s * 16 + l15;
        float bs = bias ? bias[col] : 0.0f;
#pragma unroll
        for (int mi = 0; mi < 2; ++mi) {
#pragma unroll
            for (int j = 0; j < 8; ++j) {
                float val = acc[mi][s][j] + bs;
                size_t idx = (size_t)(mb * 32 + mi * 16 + half * 8 + j) * ldc + col;
                if (OUT_BF16) ((__bf16*)C)[idx] = f2bf(val);
                else          ((float*)C)[idx]  = val;
            }
        }
    }
}

// ---------------- per-step attention score + softmax ----------------
// score[n,t] = sum_k tanh(keys[n,t,k] + q[n,k]) * v[k] + bv ; w = softmax_t(score)
// Lane L owns the contiguous K-chunk [L*32, L*32+32) -> b128 loads of bf16 keys.
__global__ __launch_bounds__(256) void attn_score_softmax(
    const __bf16* __restrict__ keysb, const float* __restrict__ q,
    const float* __restrict__ vw, const float* __restrict__ bvp,
    float* __restrict__ wout) {
    __shared__ float sc[TENC];
    __shared__ float red[8];
    int n = blockIdx.x;
    int tid = threadIdx.x, lane = tid & 31, wave = tid >> 5;

    float qreg[32], vreg[32];
    const float* qn = q + n * DM + lane * 32;
    const float* vn = vw + lane * 32;
#pragma unroll
    for (int i = 0; i < 32; ++i) { qreg[i] = qn[i]; vreg[i] = vn[i]; }
    const __bf16* kn = keysb + (size_t)n * TENC * DM + lane * 32;
    float bv0 = bvp[0];

    for (int t = wave; t < TENC; t += 8) {
        const __bf16* row = kn + (size_t)t * DM;
        __builtin_prefetch(row + (size_t)8 * DM, 0, 0);   // next t for this wave
        float s = 0.f;
#pragma unroll
        for (int c = 0; c < 4; ++c) {
            v8bf kv = *(const v8bf*)(row + c * 8);
#pragma unroll
            for (int i = 0; i < 8; ++i)
                s += tanhf(bf2f(kv[i]) + qreg[c * 8 + i]) * vreg[c * 8 + i];
        }
#pragma unroll
        for (int off = 16; off; off >>= 1) s += __shfl_down(s, off, 32);
        if (lane == 0) sc[t] = s + bv0;
    }
    __syncthreads();

    // block softmax over sc[0..TENC)
    float mx = -1e30f;
    for (int t = tid; t < TENC; t += 256) mx = fmaxf(mx, sc[t]);
#pragma unroll
    for (int off = 16; off; off >>= 1) mx = fmaxf(mx, __shfl_down(mx, off, 32));
    if (lane == 0) red[wave] = mx;
    __syncthreads();
    if (tid == 0) { float r = red[0]; for (int i = 1; i < 8; ++i) r = fmaxf(r, red[i]); red[0] = r; }
    __syncthreads();
    mx = red[0];
    __syncthreads();

    float sum = 0.f;
    for (int t = tid; t < TENC; t += 256) { float e = __expf(sc[t] - mx); sc[t] = e; sum += e; }
#pragma unroll
    for (int off = 16; off; off >>= 1) sum += __shfl_down(sum, off, 32);
    if (lane == 0) red[wave] = sum;
    __syncthreads();
    if (tid == 0) { float r = 0.f; for (int i = 0; i < 8; ++i) r += red[i]; red[0] = r; }
    __syncthreads();
    float inv = 1.0f / red[0];
    for (int t = tid; t < TENC; t += 256) wout[n * TENC + t] = sc[t] * inv;
}

// ---------------- per-step context = sum_t w[n,t] * x[n,t,:] ----------------
// Thread owns 4 columns; softmax weights staged into LDS via the Tensor Data
// Mover (2KB 1-D tile descriptor) when available.
__global__ __launch_bounds__(256) void context_kernel(const __bf16* __restrict__ xb,
                                                      const float* __restrict__ w,
                                                      __bf16* __restrict__ ctxb) {
    __shared__ float wsm[TENC];
    int n = blockIdx.x;
    int tid = threadIdx.x;
#if defined(HAVE_TDM)
    if (tid < 32) {   // one wave issues the TDM op (EXEC is ignored by TDM)
        unsigned long long ga = (unsigned long long)(const void*)(w + n * TENC);
        u32x4 g0;
        g0[0] = 1u;                                   // count=1, normal (non-restore)
        g0[1] = (unsigned)(size_t)(void*)&wsm[0];     // lds_addr = addr[31:0]
        g0[2] = (unsigned)(ga & 0xFFFFFFFFu);         // global_addr[31:0]
        g0[3] = (unsigned)((ga >> 32) & 0x1FFFFFFu) | 0x80000000u; // addr[56:32] | type=2
        i32x8 g1 = {};
        g1[0] = 0x20000;                              // data_size=2 (4 bytes)
        g1[1] = (int)((unsigned)TENC << 16);          // tensor_dim0 = 500 (lo16)
        g1[2] = (int)(1u << 16);                      // tensor_dim0 hi=0, tensor_dim1=1
        g1[3] = (int)((unsigned)TENC << 16);          // tile_dim0 = 500
        g1[5] = TENC;                                 // tensor_dim0_stride = 500
        i32x4 gz = {};
#if defined(TDM_6ARG)
        i32x8 gz8 = {};
        __builtin_amdgcn_tensor_load_to_lds(g0, g1, gz, gz, gz8, 0);
#else
        __builtin_amdgcn_tensor_load_to_lds(g0, g1, gz, gz, 0);
#endif
        __builtin_amdgcn_s_wait_tensorcnt((short)0);
    }
#else
    for (int t = tid; t < TENC; t += 256) wsm[t] = w[n * TENC + t];
#endif
    __syncthreads();

    const __bf16* xn = xb + (size_t)n * TENC * DD + tid * 4;
    float a0 = 0.f, a1 = 0.f, a2 = 0.f, a3 = 0.f;
#pragma unroll 4
    for (int t = 0; t < TENC; ++t) {
        __builtin_prefetch(xn + (size_t)(t + 16) * DD, 0, 0);
        v4bf xv = *(const v4bf*)(xn + (size_t)t * DD);
        float wv = wsm[t];
        a0 += wv * bf2f(xv[0]);
        a1 += wv * bf2f(xv[1]);
        a2 += wv * bf2f(xv[2]);
        a3 += wv * bf2f(xv[3]);
    }
    __bf16* cp = ctxb + n * DD + tid * 4;
    cp[0] = f2bf(a0); cp[1] = f2bf(a1); cp[2] = f2bf(a2); cp[3] = f2bf(a3);
}

// ---------------- per-step GRU gates (h_prev==0 for GRU internals) ----------------
__global__ __launch_bounds__(256) void gru_gates(const float* __restrict__ gx,
                                                 const float* __restrict__ gy,
                                                 const float* __restrict__ b_rec,
                                                 __bf16* __restrict__ hb,
                                                 __bf16* __restrict__ Hb, int t) {
    int idx = blockIdx.x * 256 + threadIdx.x;      // [0, NB*DM)
    int n = idx >> 10, k = idx & 1023;
    const float* gxn = gx + n * (3 * DM);
    const float* gyn = gy + (size_t)(n * TDEC + t) * (3 * DM);
    float xz = gxn[k]          + gyn[k];
    float xr = gxn[DM + k]     + gyn[DM + k];
    float xh = gxn[2 * DM + k] + gyn[2 * DM + k];
    float z  = 1.f / (1.f + __expf(-(xz + b_rec[k])));
    float r  = 1.f / (1.f + __expf(-(xr + b_rec[DM + k])));
    float hh = tanhf(xh + r * b_rec[2 * DM + k]);
    float h  = (1.f - z) * hh;
    __bf16 hv = f2bf(h);
    hb[idx] = hv;                                   // query for next step's q-GEMM
    Hb[(size_t)(n * TDEC + t) * DM + k] = hv;       // row n*TDEC+t of final out-GEMM
}

// ---------------- host orchestration ----------------
extern "C" void kernel_launch(void* const* d_in, const int* in_sizes, int n_in,
                              void* d_out, int out_size, void* d_ws, size_t ws_size,
                              hipStream_t stream) {
    (void)in_sizes; (void)n_in; (void)out_size; (void)ws_size;
    const float* x    = (const float*)d_in[0];
    const float* m    = (const float*)d_in[1];
    const int*   y    = (const int*)d_in[2];
    const float* emb  = (const float*)d_in[3];
    const float* W1   = (const float*)d_in[4];
    const float* b1   = (const float*)d_in[5];
    const float* W2   = (const float*)d_in[6];
    const float* b2   = (const float*)d_in[7];
    const float* vv   = (const float*)d_in[8];
    const float* bv   = (const float*)d_in[9];
    const float* Wx   = (const float*)d_in[10];
    /* d_in[11] = Uh: provably unused (GRU h_prev == 0) */
    const float* b_in = (const float*)d_in[12];
    const float* b_rec= (const float*)d_in[13];
    const float* Wo   = (const float*)d_in[14];
    const float* bo   = (const float*)d_in[15];
    float* out = (float*)d_out;

    char* p = (char*)d_ws;
    auto alloc = [&](size_t bytes) -> void* {
        void* r = (void*)p; p += (bytes + 255) & ~(size_t)255; return r;
    };
    __bf16* xb     = (__bf16*)alloc((size_t)NB * TENC * DD * 2);     // x in bf16 (L2-resident)
    __bf16* keysb  = (__bf16*)alloc((size_t)NB * TENC * DM * 2);     // keys in bf16
    __bf16* W1sw   = (__bf16*)alloc((size_t)DD * DM * 2);
    __bf16* W2sw   = (__bf16*)alloc((size_t)DM * DM * 2);
    __bf16* Wxcsw  = (__bf16*)alloc((size_t)DD * 3 * DM * 2);        // context rows of Wx
    __bf16* Wxesw  = (__bf16*)alloc((size_t)EE * 3 * DM * 2);        // embedding rows of Wx
    __bf16* Wosw   = (__bf16*)alloc((size_t)DM * CC * 2);
    __bf16* yembb  = (__bf16*)alloc((size_t)NB * TDEC * EE * 2);
    float*  gy     = (float*)alloc((size_t)NB * TDEC * 3 * DM * 4);  // y_emb@Wx_e + b_in
    __bf16* hb     = (__bf16*)alloc((size_t)NB * DM * 2);            // current query (bf16)
    float*  qbuf   = (float*)alloc((size_t)NB * DM * 4);
    float*  wbuf   = (float*)alloc((size_t)NB * TENC * 4);
    __bf16* ctxb   = (__bf16*)alloc((size_t)NB * DD * 2);
    float*  gxbuf  = (float*)alloc((size_t)NB * 3 * DM * 4);
    __bf16* Hb     = (__bf16*)alloc((size_t)NB * TDEC * DM * 2);

    auto gemm_blocks = [](int mblk, int nblk) { return (mblk * nblk + 7) / 8; };

    // ---- one-time precompute ----
    cvt_f32_bf16<<<(NB * TENC * DD + 255) / 256, 256, 0, stream>>>(x, xb, NB * TENC * DD);
    cvt_f32_bf16<<<(NB * DM + 255) / 256, 256, 0, stream>>>(m, hb, NB * DM);
    pack_b<<<((DD / 32) * (DM / 16) * 32 + 255) / 256, 256, 0, stream>>>(W1, W1sw, DD, DM);
    pack_b<<<((DM / 32) * (DM / 16) * 32 + 255) / 256, 256, 0, stream>>>(W2, W2sw, DM, DM);
    pack_b<<<((DD / 32) * (3 * DM / 16) * 32 + 255) / 256, 256, 0, stream>>>(Wx, Wxcsw, DD, 3 * DM);
    pack_b<<<((EE / 32) * (3 * DM / 16) * 32 + 255) / 256, 256, 0, stream>>>(Wx + (size_t)DD * 3 * DM, Wxesw, EE, 3 * DM);
    pack_b<<<((DM / 32) * (CC / 16) * 32 + 255) / 256, 256, 0, stream>>>(Wo, Wosw, DM, CC);
    gather_emb<<<(NB * TDEC * EE + 255) / 256, 256, 0, stream>>>(emb, y, yembb);

    // keys = x @ W1 + b1   [16000,1024] -> bf16
    wmma_gemm_kernel<true><<<gemm_blocks(NB * TENC / 32, DM / 64), 256, 0, stream>>>(
        xb, W1sw, b1, keysb, NB * TENC / 32, DM / 64, DD / 32, DD, DM);
    // gy = y_emb @ Wx_e + b_in   [3200,3072] f32 (hoisted across all steps)
    wmma_gemm_kernel<false><<<gemm_blocks(NB * TDEC / 32, 3 * DM / 64), 256, 0, stream>>>(
        yembb, Wxesw, b_in, gy, NB * TDEC / 32, 3 * DM / 64, EE / 32, EE, 3 * DM);

    // ---- sequential decoder steps ----
    for (int t = 0; t < TDEC; ++t) {
        // q = h @ W2 + b2   [32,1024]
        wmma_gemm_kernel<false><<<gemm_blocks(NB / 32, DM / 64), 256, 0, stream>>>(
            hb, W2sw, b2, qbuf, NB / 32, DM / 64, DM / 32, DM, DM);
        // score + softmax -> w   [32,500]
        attn_score_softmax<<<NB, 256, 0, stream>>>(keysb, qbuf, vv, bv, wbuf);
        // context = w @ x   [32,1024] bf16
        context_kernel<<<NB, 256, 0, stream>>>(xb, wbuf, ctxb);
        // gx = context @ Wx_c   [32,3072]
        wmma_gemm_kernel<false><<<gemm_blocks(NB / 32, 3 * DM / 64), 256, 0, stream>>>(
            ctxb, Wxcsw, nullptr, gxbuf, NB / 32, 3 * DM / 64, DD / 32, DD, 3 * DM);
        // gates -> h (feeds next step's query + batched output GEMM)
        gru_gates<<<(NB * DM + 255) / 256, 256, 0, stream>>>(gxbuf, gy, b_rec, hb, Hb, t);
    }

    // out = H @ Wo + bo   [3200,8000] -> d_out (layout [N, T_dec, C], row = n*TDEC+t)
    wmma_gemm_kernel<false><<<gemm_blocks(NB * TDEC / 32, CC / 64), 256, 0, stream>>>(
        Hb, Wosw, bo, out, NB * TDEC / 32, CC / 64, DM / 32, DM, CC);
}
